// EnergyBalanceChecker_47236050321912
// MI455X (gfx1250) — compile-verified, matrix operand order checked by reading, not run on hardware
//
#include <hip/hip_runtime.h>
#include <hip/hip_bf16.h>

// Problem constants (match reference)
#define BB 4
#define NN 512
#define TT 24
#define GG 32
#define BT (BB * TT)          // 96 columns after flattening (b,t)
#define APAD 514              // padded LDS row stride: 514 % 64 == 2 -> conflict-free
static constexpr float TOL = 0.05f;
static constexpr float EPS = 1e-6f;

typedef float     v2f  __attribute__((ext_vector_type(2)));
typedef float     v8f  __attribute__((ext_vector_type(8)));
typedef _Float16  v16h __attribute__((ext_vector_type(16)));

// ---------------------------------------------------------------------------
// Kernel 0: zero the scalar accumulator in workspace.
// ---------------------------------------------------------------------------
__global__ void eb_init_acc(float* acc) {
    if (threadIdx.x == 0) acc[0] = 0.0f;
}

// ---------------------------------------------------------------------------
// Kernel 1: streaming pass over the 100 MB sharing matrix (bandwidth-bound:
// ~4.3us/pass at 23.3 TB/s; second pass is L2-resident since 100MB < 192MB L2).
//   X[b,n,t] = C - G + rowsum_m S[b,n,m,t] - colsum_m S[b,m,n,t]
//   E[b,n,t] = C + G
// One thread per (b,n,t); t fastest -> coalesced across lanes. 1536 waves.
// ---------------------------------------------------------------------------
__global__ void eb_prep(const float* __restrict__ C, const float* __restrict__ Gn,
                        const float* __restrict__ S,
                        float* __restrict__ X, float* __restrict__ E) {
    int idx = blockIdx.x * blockDim.x + threadIdx.x;
    if (idx >= BB * NN * TT) return;
    int t = idx % TT;
    int n = (idx / TT) % NN;
    int b = idx / (TT * NN);

    const float* Sb = S + (size_t)b * NN * NN * TT;
    const float* pr = Sb + (size_t)n * NN * TT + t;   // row:  S[b,n,m,t], m varies
    const float* pc = Sb + (size_t)n * TT + t;        // col:  S[b,m,n,t], m varies
    float rs = 0.0f, cs = 0.0f;
    for (int m = 0; m < NN; ++m) {
        rs += pr[(size_t)m * TT];
        cs += pc[(size_t)m * NN * TT];
    }
    float c = C[idx], g = Gn[idx];
    X[idx] = (c - g) + (rs - cs);
    E[idx] = c + g;
}

// ---------------------------------------------------------------------------
// Kernel 2: WMMA matmuls  P = M @ X, Q = M @ E  fused with penalty reduction.
// Grid: 12 blocks (2 G-tiles x 6 column-tiles), 32 threads = 1 wave each,
// EXEC all-1s as WMMA requires.
//
// The block's 16 rows of M are pre-masked into an LDS A-panel so the hot loop
// contains no compares / no predicated loads:
//   sA[r*APAD + k] = (ids[k] == tg*16 + r) ? msk[k] : 0
// A-fragment (f32 16x4 layout: lanes0-15 {K0,K1}, lanes16-31 {K2,K3}) is then
// a single aligned ds_load_b64 per step.
// ---------------------------------------------------------------------------
__global__ void eb_wmma_penalty(const float* __restrict__ X, const float* __restrict__ E,
                                const int* __restrict__ ids, const float* __restrict__ msk,
                                float* __restrict__ acc) {
    __shared__ float sA[16 * APAD];               // ~33 KB of the 320 KB WGP LDS

    const int lane = threadIdx.x;                 // 0..31
    const int tile = blockIdx.x;                  // 0..11
    const int tg   = tile & 1;                    // which 16-row block of G
    const int tc   = tile >> 1;                   // which 16-col block of B*T

    // Build the masked A-panel: 8192 elements / 32 lanes = 256 iterations.
    // Loads are unconditional (locals), select lowers to v_cndmask.
    for (int i = lane; i < 16 * NN; i += 32) {
        int r  = i >> 9;                          // 0..15
        int k  = i & (NN - 1);                    // 0..511
        int id = ids[k];
        float m = msk[k];
        sA[r * APAD + k] = (id == tg * 16 + r) ? m : 0.0f;
    }
    __syncthreads();

    const int r    = lane & 15;
    const int hi   = lane >> 4;                   // lanes 16-31 carry K+2 (f32 layout)
    const int col  = tc * 16 + r;                 // output column this lane supplies for B
    const int cb   = col / TT, ct = col % TT;     // (b,t) of that column
    const float* Xc = X + (size_t)cb * NN * TT + ct;   // + n*TT walks down the column
    const float* Ec = E + (size_t)cb * NN * TT + ct;
    const v2f* __restrict__ pa = (const v2f*)(sA + r * APAD);  // row base, 8B aligned

    v8f accP = {0.f, 0.f, 0.f, 0.f, 0.f, 0.f, 0.f, 0.f};
    v8f accQ = {0.f, 0.f, 0.f, 0.f, 0.f, 0.f, 0.f, 0.f};

#if __has_builtin(__builtin_amdgcn_wmma_f32_16x16x4_f32)
    // Exact fp32 path: V_WMMA_F32_16X16X4_F32, K-loop over 512 in steps of 4.
#pragma unroll 4
    for (int k = 0; k < NN; k += 4) {
        const int ka = k + (hi ? 2 : 0);          // even -> pa[ka>>1] is one b64
        v2f a = pa[ka >> 1];
        v2f bx, be;
        bx.x = Xc[(size_t)ka * TT];
        bx.y = Xc[(size_t)(ka + 1) * TT];
        be.x = Ec[(size_t)ka * TT];
        be.y = Ec[(size_t)(ka + 1) * TT];
        accP = __builtin_amdgcn_wmma_f32_16x16x4_f32(false, a, false, bx,
                                                     (short)0, accP, false, false);
        accQ = __builtin_amdgcn_wmma_f32_16x16x4_f32(false, a, false, be,
                                                     (short)0, accQ, false, false);
    }
#else
    // Fallback: codegen-confirmed V_WMMA_F32_16X16X32_F16, K step 32.
    for (int k = 0; k < NN; k += 32) {
        v16h a, bx, be;
#pragma unroll
        for (int i = 0; i < 16; ++i) {
            int kk = k + ((i < 8) ? i : 16 + (i - 8)) + (hi ? 8 : 0);
            a[i]  = (_Float16)sA[r * APAD + kk];
            bx[i] = (_Float16)Xc[(size_t)kk * TT];
            be[i] = (_Float16)Ec[(size_t)kk * TT];
        }
        accP = __builtin_amdgcn_wmma_f32_16x16x32_f16(false, a, false, bx,
                                                      (short)0, accP, false, false);
        accQ = __builtin_amdgcn_wmma_f32_16x16x32_f16(false, a, false, be,
                                                      (short)0, accQ, false, false);
    }
#endif

    // Penalty: relu(|P| / (Q + eps) - tol), summed over this tile.
    // Q >= eps > 0 and the result feeds a 0.05-threshold relu, so a single
    // v_rcp_f32 (~1 ulp) replaces the ~10-op IEEE divide expansion.
    float s = 0.0f;
#pragma unroll
    for (int j = 0; j < 8; ++j) {
        float p   = fabsf(accP[j]);
        float q   = accQ[j] + EPS;
        float rel = p * __builtin_amdgcn_rcpf(q) - TOL;
        s += (rel > 0.0f) ? rel : 0.0f;
    }
    // wave32 butterfly reduction
    for (int off = 16; off > 0; off >>= 1)
        s += __shfl_xor(s, off, 32);
    if (lane == 0) atomicAdd(acc, s);
}

// ---------------------------------------------------------------------------
// Kernel 3: n_unique via per-lane group presence + ballot; final scale.
// ---------------------------------------------------------------------------
__global__ void eb_finalize(const int* __restrict__ ids, const float* __restrict__ w,
                            const float* __restrict__ acc, float* __restrict__ out) {
    const int g = threadIdx.x;                    // 0..31 == group id range
    bool present = false;
    for (int n = 0; n < NN; ++n) present = present || (ids[n] == g);
    unsigned long long bal = __ballot(present);
    if (g == 0) {
        float nuniq = (float)__popcll(bal);
        out[0] = acc[0] * w[0] / nuniq;
    }
}

// ---------------------------------------------------------------------------
extern "C" void kernel_launch(void* const* d_in, const int* in_sizes, int n_in,
                              void* d_out, int out_size, void* d_ws, size_t ws_size,
                              hipStream_t stream) {
    const float* C   = (const float*)d_in[0];   // consumption [B,N,T]
    const float* Gn  = (const float*)d_in[1];   // generation  [B,N,T]
    const float* S   = (const float*)d_in[2];   // sharing     [B,N,N,T]
    const int*   ids = (const int*)  d_in[3];   // lv_group_ids [N]
    const float* msk = (const float*)d_in[4];   // valid_lv_mask [N]
    const float* w   = (const float*)d_in[5];   // imbalance_penalty_weight [1]
    float* out = (float*)d_out;

    float* ws  = (float*)d_ws;
    float* acc = ws;                      // 1 float (16-float slot for alignment)
    float* X   = ws + 16;                 // [B,N,T] = 49152 floats
    float* E   = X + BB * NN * TT;        // [B,N,T] = 49152 floats

    eb_init_acc<<<1, 32, 0, stream>>>(acc);

    int nElem = BB * NN * TT;             // 49152
    eb_prep<<<(nElem + 255) / 256, 256, 0, stream>>>(C, Gn, S, X, E);

    eb_wmma_penalty<<<12, 32, 0, stream>>>(X, E, ids, msk, acc);

    eb_finalize<<<1, 32, 0, stream>>>(ids, w, acc, out);
}
